// MultiHeadAttention_12292196401937
// MI455X (gfx1250) — matrix operand order repaired and checked by reading, not compiled, not run on hardware
//
#include <hip/hip_runtime.h>
#include <hip/hip_bf16.h>

typedef __attribute__((ext_vector_type(16))) _Float16 v16h;
typedef __attribute__((ext_vector_type(8)))  _Float16 v8h;
typedef __attribute__((ext_vector_type(2)))  _Float16 h2v;
typedef __attribute__((ext_vector_type(8)))  float    v8f;
typedef __attribute__((ext_vector_type(4)))  float    v4f;

// ---------------------------------------------------------------------------
// WMMA fragment loaders (CDNA5 16x16x32 f16 layouts, wave32).
// A (16x32 MxK): lane r=lane&15, kb=(lane>=16)?8:0; two 8-half chunks at
//                k=kb..kb+7 and k=kb+16..kb+23.
// B (32x16 KxN) staged transposed Bt[n][k]: lane n=lane&15, kb=(lane>=16)?8:0;
//                two 8-half chunks at k=kb..kb+7 and k=kb+16..kb+23.
// C/D: vgpr j, lane -> row j+((lane>=16)?8:0), col lane&15.
// ---------------------------------------------------------------------------
__device__ __forceinline__ v16h frag_a(const _Float16* base, int stride, int lane) {
  const int r  = lane & 15;
  const int kb = (lane >> 4) << 3;
  const _Float16* p = base + r * stride;
  v8h a0 = *(const v8h*)(p + kb);
  v8h a1 = *(const v8h*)(p + kb + 16);
  v16h f;
#pragma unroll
  for (int i = 0; i < 8; ++i) { f[i] = a0[i]; f[i + 8] = a1[i]; }
  return f;
}

__device__ __forceinline__ v16h frag_b(const _Float16* base, int stride, int lane) {
  const int n  = lane & 15;
  const int kb = (lane >> 4) << 3;
  const _Float16* p = base + n * stride + kb;
  v8h b0 = *(const v8h*)(p);
  v8h b1 = *(const v8h*)(p + 16);
  v16h f;
#pragma unroll
  for (int i = 0; i < 8; ++i) { f[i] = b0[i]; f[i + 8] = b1[i]; }
  return f;
}

__device__ __forceinline__ v8f wmma_f16(v16h a, v16h b, v8f c) {
  return __builtin_amdgcn_wmma_f32_16x16x32_f16(false, a, false, b,
                                                (short)0, c, false, false);
}

// --- CDNA5 async global->LDS copy (16 bytes per lane), ASYNCcnt-tracked -----
__device__ __forceinline__ void async_b128(void* lds, const void* gptr) {
  unsigned lds_off = (unsigned)(uintptr_t)lds;           // low32 of shared flat
  unsigned long long ga = (unsigned long long)(uintptr_t)gptr;
  asm volatile("global_load_async_to_lds_b128 %0, %1, off"
               :: "v"(lds_off), "v"(ga) : "memory");
}
__device__ __forceinline__ void wait_async0() {
  asm volatile("s_wait_asynccnt 0x0" ::: "memory");
}

// ---------------------------------------------------------------------------
// Kernel 1: fused QKV projection, 64(M)x128(N) tile, k-step 32, 8 waves.
// Each wave: 1 A-frag reused over 4 B-frags -> 4 WMMAs/step.
//   which=0: qh [b][h][s][dk]   which=1: kh [b][h][s][dk]
//   which=2: vh TRANSPOSED [b][h][dv][s]
// ---------------------------------------------------------------------------
__global__ __launch_bounds__(256) void mha_proj_kernel(
    const float* __restrict__ qin, const float* __restrict__ kin,
    const float* __restrict__ vin,
    const float* __restrict__ wq, const float* __restrict__ bq,
    const float* __restrict__ wk, const float* __restrict__ bk,
    const float* __restrict__ wv, const float* __restrict__ bv,
    _Float16* __restrict__ qh, _Float16* __restrict__ kh,
    _Float16* __restrict__ vht)
{
  constexpr int LDA = 40;                       // mult of 8 halves (16B align)
  constexpr int LDB = 40;
  __shared__ __align__(16) _Float16 sA [64  * LDA];
  __shared__ __align__(16) _Float16 sBt[128 * LDB];     // Bt[n][k]

  const int tid = threadIdx.x, lane = tid & 31, w = tid >> 5;
  const int m0 = (blockIdx.x >> 3) * 64;
  const int n0 = (blockIdx.x & 7) * 128;
  const int which = blockIdx.y;

  const float* X  = (which == 0) ? qin : (which == 1) ? kin : vin;
  const float* W  = (which == 0) ? wq  : (which == 1) ? wk  : wv;
  const float* Bi = (which == 0) ? bq  : (which == 1) ? bk  : bv;

  const int qi    = w >> 1;                     // M-subtile 0..3
  const int nhalf = (w & 1) * 64;               // N half within 128
  v8f acc[4] = {{}, {}, {}, {}};

  const int ar = tid >> 2, ac0 = (tid & 3) * 8;           // A staging coords

  for (int k0 = 0; k0 < 1024; k0 += 32) {
    { // stage A 64x32: two float4 loads -> one b128 LDS store per thread
      const float* src = X + (size_t)(m0 + ar) * 1024 + k0 + ac0;
      v4f x0 = *(const v4f*)(src);
      v4f x1 = *(const v4f*)(src + 4);
      v8h hv;
#pragma unroll
      for (int i = 0; i < 4; ++i) { hv[i] = (_Float16)x0[i]; hv[i+4] = (_Float16)x1[i]; }
      *(v8h*)(sA + ar * LDA + ac0) = hv;
    }
    // stage Bt[n][k]=W[k][n], packing k-adjacent pairs into b32 stores
#pragma unroll
    for (int i = 0; i < 8; ++i) {
      int pid = tid + i * 256;                  // 0..2047
      int n = pid & 127, cp = pid >> 7;         // n-fast (coalesced), cp 0..15
      const float* wp = W + (size_t)(k0 + 2 * cp) * 1024 + n0 + n;
      h2v pk;
      pk[0] = (_Float16)wp[0];
      pk[1] = (_Float16)wp[1024];
      *(h2v*)(sBt + n * LDB + 2 * cp) = pk;
    }
    __syncthreads();

    v16h a = frag_a(sA + qi * 16 * LDA, LDA, lane);
#pragma unroll
    for (int t = 0; t < 4; ++t) {
      v16h bt = frag_b(sBt + (nhalf + t * 16) * LDB, LDB, lane);
      acc[t] = wmma_f16(a, bt, acc[t]);
    }
    __syncthreads();
  }

  // Epilogue: bias + scatter to head layout (f16)
  const int msub = (lane >> 4) << 3;
#pragma unroll
  for (int t = 0; t < 4; ++t) {
    const int ncol = n0 + nhalf + t * 16 + (lane & 15);
    const float bias = Bi[ncol];
    const int hh = ncol >> 6, dd = ncol & 63;
#pragma unroll
    for (int j = 0; j < 8; ++j) {
      int m = m0 + qi * 16 + j + msub;
      int b = m >> 11, s = m & 2047;
      float val = acc[t][j] + bias;
      if (which == 2)
        vht[((size_t)(b * 16 + hh) * 64 + dd) * 2048 + s] = (_Float16)val;
      else {
        _Float16* O = (which == 0) ? qh : kh;
        O[((size_t)(b * 16 + hh) * 2048 + s) * 64 + dd] = (_Float16)val;
      }
    }
  }
}

// ---------------------------------------------------------------------------
// Kernel 2: flash attention per (b, h, 64-query tile), 8 waves.
// K/V/Q tiles staged with async global->LDS b128; parallel online softmax.
// Reference scale: scores / DK**(-0.5) == scores * 8.
// ---------------------------------------------------------------------------
__global__ __launch_bounds__(256) void mha_attn_kernel(
    const _Float16* __restrict__ qh, const _Float16* __restrict__ kh,
    const _Float16* __restrict__ vht, const int* __restrict__ mask,
    const float* __restrict__ resid, float* __restrict__ out)
{
  constexpr int LD = 72;
  constexpr int LS = 66;
  __shared__ __align__(16) _Float16 Qs [64 * LD];
  __shared__ __align__(16) _Float16 Ks [64 * LD];
  __shared__ __align__(16) _Float16 Vts[64 * LD];       // Vt[d][kk]
  __shared__ __align__(16) _Float16 Ps [64 * LD];       // probs f16
  __shared__ float Sb[64 * LS];
  __shared__ float pred[256];                           // partial max/sum
  __shared__ float mrow[64], lrow[64], arow[64];
  __shared__ int   mcol[64];

  const int tid = threadIdx.x, lane = tid & 31, w = tid >> 5;
  const int qt = blockIdx.x & 31;
  const int h  = (blockIdx.x >> 5) & 15;
  const int b  = blockIdx.x >> 9;

  const size_t headbase = (size_t)(b * 16 + h) * 2048 * 64;
  const _Float16* Qg = qh  + headbase + (size_t)qt * 64 * 64;
  const _Float16* Kg = kh  + headbase;
  const _Float16* Vg = vht + headbase;                  // [d][s]

  const int sr = tid >> 2, sc0 = (tid & 3) * 16;        // staging coords
  // stage Q tile once (async, 2 x 16B per thread)
  async_b128(Qs + sr * LD + sc0,     Qg + sr * 64 + sc0);
  async_b128(Qs + sr * LD + sc0 + 8, Qg + sr * 64 + sc0 + 8);
  if (tid < 64) { mrow[tid] = -1e30f; lrow[tid] = 0.0f; }

  const int qi   = w >> 1;
  const int half = w & 1;
  const int rbase = qi * 16 + ((lane >> 4) << 3);
  v8f o0 = {}, o1 = {};
  wait_async0();
  __syncthreads();

  for (int kb = 0; kb < 32; ++kb) {
    // --- async stage K[kk][d] and Vt[d][kk] tiles
    {
      const _Float16* kgp = Kg + (size_t)(kb * 64 + sr) * 64 + sc0;
      async_b128(Ks + sr * LD + sc0,     kgp);
      async_b128(Ks + sr * LD + sc0 + 8, kgp + 8);
      const _Float16* vgp = Vg + (size_t)sr * 2048 + kb * 64 + sc0;
      async_b128(Vts + sr * LD + sc0,     vgp);
      async_b128(Vts + sr * LD + sc0 + 8, vgp + 8);
    }
    if (tid < 64) mcol[tid] = mask[b * 2048 + kb * 64 + tid];
    wait_async0();
    __syncthreads();

    // --- scores: S = Q @ K^T (wave owns subtiles (qi, half*2..half*2+1))
    v16h aq0 = frag_a(Qs + qi * 16 * LD,      LD, lane);
    v16h aq1 = frag_a(Qs + qi * 16 * LD + 32, LD, lane);
#pragma unroll
    for (int t = 0; t < 2; ++t) {
      int kj = half * 2 + t;
      v16h bk0 = frag_b(Ks + kj * 16 * LD,      LD, lane);
      v16h bk1 = frag_b(Ks + kj * 16 * LD + 32, LD, lane);
      v8f sc = {};
      sc = wmma_f16(aq0, bk0, sc);
      sc = wmma_f16(aq1, bk1, sc);
      int col = kj * 16 + (lane & 15);
#pragma unroll
      for (int j = 0; j < 8; ++j) Sb[(rbase + j) * LS + col] = sc[j];
    }
    __syncthreads();

    // --- parallel online softmax: 4 threads per row, 16 cols each
    {   // P1: scale + mask + partial max
      int r = tid >> 2, c0 = (tid & 3) * 16;
      float lmax = -1e30f;
#pragma unroll
      for (int c = c0; c < c0 + 16; ++c) {
        float s = Sb[r * LS + c] * 8.0f;
        if (mcol[c] == 0) s = -1e9f;
        Sb[r * LS + c] = s;
        lmax = fmaxf(lmax, s);
      }
      pred[tid] = lmax;
    }
    __syncthreads();
    if (tid < 64) {  // P2: row max + alpha
      float mold = mrow[tid];
      float mmax = fmaxf(fmaxf(pred[tid*4], pred[tid*4+1]),
                         fmaxf(pred[tid*4+2], fmaxf(pred[tid*4+3], mold)));
      arow[tid] = __expf(mold - mmax);
      mrow[tid] = mmax;
    }
    __syncthreads();
    {   // P3: exp + probs f16 + partial sum
      int r = tid >> 2, c0 = (tid & 3) * 16;
      float mm = mrow[r], sum = 0.0f;
#pragma unroll
      for (int c = c0; c < c0 + 16; ++c) {
        float p = __expf(Sb[r * LS + c] - mm);
        Ps[r * LD + c] = (_Float16)p;
        sum += p;
      }
      pred[tid] = sum;
    }
    __syncthreads();
    if (tid < 64)    // P4: running denominator
      lrow[tid] = lrow[tid] * arow[tid] +
                  (pred[tid*4] + pred[tid*4+1] + pred[tid*4+2] + pred[tid*4+3]);

    // --- rescale accumulators by alpha(row)
#pragma unroll
    for (int j = 0; j < 8; ++j) {
      float al = arow[rbase + j];
      o0[j] *= al; o1[j] *= al;
    }
    // --- O += P @ V (B operand = Vt rows, contiguous kk)
    v16h ap0 = frag_a(Ps + qi * 16 * LD,      LD, lane);
    v16h ap1 = frag_a(Ps + qi * 16 * LD + 32, LD, lane);
    {
      int dj = half * 2;
      v16h bv00 = frag_b(Vts + dj * 16 * LD,            LD, lane);
      v16h bv01 = frag_b(Vts + dj * 16 * LD + 32,       LD, lane);
      o0 = wmma_f16(ap0, bv00, o0);
      o0 = wmma_f16(ap1, bv01, o0);
      v16h bv10 = frag_b(Vts + (dj + 1) * 16 * LD,      LD, lane);
      v16h bv11 = frag_b(Vts + (dj + 1) * 16 * LD + 32, LD, lane);
      o1 = wmma_f16(ap0, bv10, o1);
      o1 = wmma_f16(ap1, bv11, o1);
    }
    __syncthreads();
  }

  // --- epilogue: normalize, add residual, store fp32
  {
    int d0 = half * 32 + (lane & 15);
#pragma unroll
    for (int j = 0; j < 8; ++j) {
      int rl = rbase + j;
      int s  = qt * 64 + rl;
      float inv = 1.0f / lrow[rl];
      size_t base = ((size_t)(b * 2048 + s)) * 1024 + h * 64;
      out[base + d0]      = o0[j] * inv + resid[base + d0];
      out[base + d0 + 16] = o1[j] * inv + resid[base + d0 + 16];
    }
  }
}

// ---------------------------------------------------------------------------
extern "C" void kernel_launch(void* const* d_in, const int* in_sizes, int n_in,
                              void* d_out, int out_size, void* d_ws, size_t ws_size,
                              hipStream_t stream) {
  (void)in_sizes; (void)n_in; (void)out_size; (void)ws_size;
  const float* q   = (const float*)d_in[0];
  const float* k   = (const float*)d_in[1];
  const float* v   = (const float*)d_in[2];
  const int*   pm  = (const int*)  d_in[3];
  const float* wq  = (const float*)d_in[4];
  const float* bq  = (const float*)d_in[5];
  const float* wk  = (const float*)d_in[6];
  const float* bk  = (const float*)d_in[7];
  const float* wv  = (const float*)d_in[8];
  const float* bv  = (const float*)d_in[9];
  float* out = (float*)d_out;

  const size_t elems = (size_t)4 * 2048 * 1024;         // B*S*H*DK
  _Float16* qh  = (_Float16*)d_ws;
  _Float16* kh  = qh + elems;
  _Float16* vht = kh + elems;                           // 48 MB, L2-resident

  dim3 gproj(128 * 8, 3);                               // Mtiles*Ntiles, {q,k,v}
  mha_proj_kernel<<<gproj, 256, 0, stream>>>(q, k, v, wq, bq, wk, bk, wv, bv,
                                             qh, kh, vht);
  mha_attn_kernel<<<4 * 16 * 32, 256, 0, stream>>>(qh, kh, vht, pm, q, out);
}
